// DecoderModule_25340307046462
// MI455X (gfx1250) — matrix operand order repaired and checked by reading, not compiled, not run on hardware
//
#include <hip/hip_runtime.h>
#include <hip/hip_bf16.h>
#include <math.h>

// Problem dims (fixed by the reference)
#define NB 32      // batch
#define TT 128     // decode steps
#define T1E 128    // encoder steps
#define HH 512     // hidden == embed
#define SS 1024    // encoder_state
#define VV 32000   // vocab
#define EPAD 514   // padded LDS row (kills bank conflicts on column reads)

typedef __attribute__((ext_vector_type(16))) __bf16 v16bf;
typedef __attribute__((ext_vector_type(8)))  __bf16 v8bf;
typedef __attribute__((ext_vector_type(8)))  float  v8f;

#define DEV static __device__ __forceinline__

DEV __bf16 f2bf(float f) {
    unsigned u = __builtin_bit_cast(unsigned, f);
    unsigned r = u + 0x7FFFu + ((u >> 16) & 1u);   // round-to-nearest-even
    unsigned short h = (unsigned short)(r >> 16);
    return __builtin_bit_cast(__bf16, h);
}
DEV float bf2f(__bf16 b) {
    unsigned short h = __builtin_bit_cast(unsigned short, b);
    unsigned u = ((unsigned)h) << 16;
    return __builtin_bit_cast(float, u);
}

// A fragment: 16x32 bf16 tile of a row-major [rows x ld] matrix at (r0, k0).
// ISA layout: lanes 0-15 -> row r0+l, K {0..7} in v0-3 and {16..23} in v4-7;
// lanes 16-31 -> same rows, K {8..15} and {24..31}.
DEV v16bf load_a_frag(const __bf16* p, int r0, int k0, int ld) {
    int l = threadIdx.x & 31;
    const __bf16* base = p + (size_t)(r0 + (l & 15)) * ld + k0 + ((l >> 4) << 3);
    v8bf lo = *(const v8bf*)(base);
    v8bf hi = *(const v8bf*)(base + 16);
    v16bf o;
#pragma unroll
    for (int i = 0; i < 8; ++i) { o[i] = lo[i]; o[i + 8] = hi[i]; }
    return o;
}

// B fragment: 32x16 bf16 (KxN), sourced from B^T stored row-major [N x ld].
// ISA layout: lane n<16 holds col n, K {0..7} and {16..23}; lane n+16 holds K {8..15} and {24..31}.
DEV v16bf load_b_frag(const __bf16* p, int c0, int k0, int ld) {
    int l = threadIdx.x & 31;
    const __bf16* base = p + (size_t)(c0 + (l & 15)) * ld + k0 + ((l >> 4) << 3);
    v8bf lo = *(const v8bf*)(base);
    v8bf hi = *(const v8bf*)(base + 16);
    v16bf o;
#pragma unroll
    for (int i = 0; i < 8; ++i) { o[i] = lo[i]; o[i + 8] = hi[i]; }
    return o;
}

DEV v8f wmma_bf16(v16bf a, v16bf b, v8f c) {
    return __builtin_amdgcn_wmma_f32_16x16x32_bf16(false, a, false, b, (short)0, c,
                                                   false, false);
}

// ---------------------------------------------------------------------------
// Generic bf16 GEMM: out[M x N] (f32) = A[M x K] * B^T  (B stored [N x K]) + bias
// Workgroup tile 64(M) x 128(N); 8 waves in 2x4 grid; wave tile 32x32 = 2x2 WMMA.
// All operands streamed straight from L2 (A, B are L2-resident for this problem).
// grid = (N/128, M/64), block = 256.
// ---------------------------------------------------------------------------
__global__ void k_gemm(const __bf16* __restrict__ A, int lda,
                       const __bf16* __restrict__ B, int ldb,
                       const float* __restrict__ bias0,
                       const float* __restrict__ bias1,
                       float* __restrict__ out, int ldo, int K) {
    const int w  = threadIdx.x >> 5;
    const int m0 = blockIdx.y * 64  + (w & 1) * 32;
    const int n0 = blockIdx.x * 128 + (w >> 1) * 32;
    v8f c00 = {}, c01 = {}, c10 = {}, c11 = {};
    for (int k0 = 0; k0 < K; k0 += 32) {
        v16bf a0 = load_a_frag(A, m0,      k0, lda);
        v16bf a1 = load_a_frag(A, m0 + 16, k0, lda);
        v16bf b0 = load_b_frag(B, n0,      k0, ldb);
        v16bf b1 = load_b_frag(B, n0 + 16, k0, ldb);
        c00 = wmma_bf16(a0, b0, c00);
        c01 = wmma_bf16(a0, b1, c01);
        c10 = wmma_bf16(a1, b0, c10);
        c11 = wmma_bf16(a1, b1, c11);
    }
    const int l  = threadIdx.x & 31;
    const int cb = l & 15;
    const int rb = (l >> 4) << 3;
    v8f cc[2][2] = {{c00, c01}, {c10, c11}};
#pragma unroll
    for (int nt = 0; nt < 2; ++nt) {
        int col = n0 + nt * 16 + cb;
        float bv = (bias0 ? bias0[col] : 0.f) + (bias1 ? bias1[col] : 0.f);
#pragma unroll
        for (int mt = 0; mt < 2; ++mt) {
#pragma unroll
            for (int i = 0; i < 8; ++i) {
                int row = m0 + mt * 16 + rb + i;
                out[(size_t)row * ldo + col] = cc[mt][nt][i] + bv;
            }
        }
    }
}

// ---------------------------------------------------------------------------
// Prep kernels
// ---------------------------------------------------------------------------

// hx0 = tanh(encoder_state @ Wp^T + bp) -> bf16 into Abuf[:, 0:512]; reset barrier.
__global__ void k_init(const float* __restrict__ es, const float* __restrict__ Wp,
                       const float* __restrict__ bp, __bf16* __restrict__ Abuf,
                       unsigned* __restrict__ bar) {
    int idx = blockIdx.x * 256 + threadIdx.x;     // 64 blocks -> 32*512
    if (idx == 0) *bar = 0u;                      // ticket barrier must start at mult of 32
    int n = idx >> 9, h = idx & (HH - 1);
    const float* e  = es + (size_t)n * SS;
    const float* wr = Wp + (size_t)h * SS;
    float s = bp[h];
    for (int k = 0; k < SS; ++k) s += e[k] * wr[k];
    Abuf[n * 2 * HH + h] = f2bf(tanhf(s));
}

// W2 = [W_hh | W_ih[:,512:1024]] bf16 [512 x 1024]; Wx = W_ih[:,0:512] bf16 [512 x 512]
__global__ void k_build_w2(const float* __restrict__ W_ih, const float* __restrict__ W_hh,
                           __bf16* __restrict__ W2, __bf16* __restrict__ Wx) {
    int idx = blockIdx.x * 256 + threadIdx.x;     // 2048 blocks -> 512*1024
    int h = idx >> 10, k = idx & 1023;
    if (k < HH) {
        W2[idx]          = f2bf(W_hh[(size_t)h * HH + k]);
        Wx[h * HH + k]   = f2bf(W_ih[(size_t)h * 2 * HH + k]);
    } else {
        W2[idx]          = f2bf(W_ih[(size_t)h * 2 * HH + k]);
    }
}

__global__ void k_cvt(const float* __restrict__ src, __bf16* __restrict__ dst, int n) {
    for (int i = blockIdx.x * blockDim.x + threadIdx.x; i < n; i += gridDim.x * blockDim.x)
        dst[i] = f2bf(src[i]);
}

// xe[r][e] = bf16(emb[x[r]][e]), r = n*T + t
__global__ void k_gather(const int* __restrict__ x, const float* __restrict__ emb,
                         __bf16* __restrict__ xe) {
    int idx = blockIdx.x * 256 + threadIdx.x;     // 8192 blocks -> 4096*512
    int r = idx >> 9, e = idx & (HH - 1);
    xe[idx] = f2bf(emb[(size_t)x[r] * HH + e]);
}

// ---------------------------------------------------------------------------
// Persistent scan kernel: 32 workgroups (one per batch row n), 256 threads.
// Phase 1 (per-n): scores/softmax/att from LDS-resident encoder slice.
// Phase 2 (per-column-group): WMMA GEMM for 16 hidden columns, all 32 rows.
// Device-wide ticket barrier between phases (32 co-resident WGs, 1 per WGP).
// ---------------------------------------------------------------------------
DEV void grid_sync(unsigned* bar) {
    __threadfence();
    __syncthreads();
    if (threadIdx.x == 0) {
        unsigned ticket = __hip_atomic_fetch_add(bar, 1u, __ATOMIC_ACQ_REL,
                                                 __HIP_MEMORY_SCOPE_AGENT);
        unsigned target = (ticket / NB + 1u) * NB;
        while (__hip_atomic_load(bar, __ATOMIC_ACQUIRE, __HIP_MEMORY_SCOPE_AGENT) < target)
            __builtin_amdgcn_s_sleep(1);
    }
    __syncthreads();
}

__global__ void k_scan(const float* __restrict__ enc,    // [32][128][512] f32
                       const float* __restrict__ xproj,  // [4096][512] f32 (biases folded)
                       const __bf16* __restrict__ W2,    // [512][1024] bf16
                       __bf16* __restrict__ Abuf,        // [32][1024] bf16 = [hx | att]
                       __bf16* __restrict__ hs,          // [4096][512] bf16
                       unsigned* __restrict__ bar) {
    extern __shared__ char smem[];
    float*  red     = (float*)smem;                  // 8 waves * 32 lanes * 8 = 2048 f32
    float*  partial = red + 2048;                    // 256 f32
    float*  alpha   = partial + 256;                 // 128 f32
    __bf16* hxl     = (__bf16*)(alpha + 128);        // 512 bf16
    __bf16* encL    = hxl + HH;                      // 128 * 514 bf16 (~131 KB)

    const int n   = blockIdx.x;
    const int tid = threadIdx.x;
    const int l   = tid & 31;
    const int w   = tid >> 5;

    // One-time: stage this batch row's encoder outputs into LDS as bf16 (padded rows).
    for (int idx = tid; idx < T1E * HH; idx += 256) {
        int t1 = idx >> 9, h = idx & (HH - 1);
        encL[t1 * EPAD + h] = f2bf(enc[((size_t)n * T1E + t1) * HH + h]);
    }
    __syncthreads();

    for (int t = 0; t < TT; ++t) {
        // ---------- phase 1: attention for batch row n ----------
        for (int h = tid; h < HH; h += 256) hxl[h] = Abuf[n * 2 * HH + h];
        __syncthreads();
        {   // scores: 2 threads per t1, each over half of H
            int t1 = tid & (T1E - 1);
            int half = tid >> 7;
            const __bf16* er = encL + t1 * EPAD + half * 256;
            const __bf16* hr = hxl + half * 256;
            float s = 0.f;
#pragma unroll 8
            for (int h = 0; h < 256; ++h) s += bf2f(er[h]) * bf2f(hr[h]);
            partial[tid] = s;
        }
        __syncthreads();
        if (w == 0) {   // softmax over 128 scores by wave 0
            float v[4];
            float mx = -1e30f;
#pragma unroll
            for (int j = 0; j < 4; ++j) {
                int t1 = l + 32 * j;
                v[j] = partial[t1] + partial[128 + t1];
                mx = fmaxf(mx, v[j]);
            }
#pragma unroll
            for (int off = 16; off > 0; off >>= 1) mx = fmaxf(mx, __shfl_xor(mx, off, 32));
            float sum = 0.f;
#pragma unroll
            for (int j = 0; j < 4; ++j) { v[j] = __expf(v[j] - mx); sum += v[j]; }
#pragma unroll
            for (int off = 16; off > 0; off >>= 1) sum += __shfl_xor(sum, off, 32);
            float inv = 1.f / sum;
#pragma unroll
            for (int j = 0; j < 4; ++j) alpha[l + 32 * j] = v[j] * inv;
        }
        __syncthreads();
        {   // att[h] = sum_t1 alpha[t1] * enc[t1][h]; write to Abuf[n][512+h]
#pragma unroll
            for (int j = 0; j < 2; ++j) {
                int h = tid + j * 256;
                float a = 0.f;
#pragma unroll 8
                for (int t1 = 0; t1 < T1E; ++t1) a += alpha[t1] * bf2f(encL[t1 * EPAD + h]);
                Abuf[n * 2 * HH + HH + h] = f2bf(a);
            }
        }
        grid_sync(bar);

        // ---------- phase 2: recurrent GEMM, this WG owns 16 hidden columns ----------
        {
            const int g  = blockIdx.x;       // column group: cols [16g, 16g+16)
            const int mt = w & 1;            // m-tile: rows 0..15 / 16..31
            const int kq = w >> 1;           // k quarter: K range [256*kq, +256)
            const int c0 = g * 16;
            v8f acc = {};
            for (int kk = 0; kk < 8; ++kk) {
                int k0 = kq * 256 + kk * 32;
                v16bf a = load_a_frag(Abuf, mt * 16, k0, 2 * HH);
                v16bf b = load_b_frag(W2,   c0,      k0, 2 * HH);
                acc = wmma_bf16(a, b, acc);
            }
            float* rslot = red + ((size_t)w * 32 + l) * 8;
#pragma unroll
            for (int i = 0; i < 8; ++i) rslot[i] = acc[i];
            __syncthreads();
            if (w < 2) {                      // reduce k-quarters, tanh, write hx/hs
                v8f c = {};
#pragma unroll
                for (int q = 0; q < 4; ++q) {
                    float* rs = red + (((size_t)(q * 2 + w)) * 32 + l) * 8;
#pragma unroll
                    for (int i = 0; i < 8; ++i) c[i] += rs[i];
                }
                int col = c0 + (l & 15);
                int rb  = (l >> 4) << 3;
#pragma unroll
                for (int i = 0; i < 8; ++i) {
                    int m = w * 16 + rb + i;  // batch row
                    float val = tanhf(c[i] + xproj[((size_t)m * TT + t) * HH + col]);
                    __bf16 bv = f2bf(val);
                    hs[((size_t)m * TT + t) * HH + col] = bv;
                    Abuf[m * 2 * HH + col] = bv;     // hx for next step
                }
            }
            __syncthreads();
        }
        grid_sync(bar);
    }
}

// ---------------------------------------------------------------------------
extern "C" void kernel_launch(void* const* d_in, const int* in_sizes, int n_in,
                              void* d_out, int out_size, void* d_ws, size_t ws_size,
                              hipStream_t stream) {
    const int*   x     = (const int*)  d_in[0];
    const float* es    = (const float*)d_in[1];
    const float* enc   = (const float*)d_in[2];
    const float* emb   = (const float*)d_in[3];
    const float* Wp    = (const float*)d_in[4];
    const float* bp    = (const float*)d_in[5];
    const float* W_ih  = (const float*)d_in[6];
    const float* b_ih  = (const float*)d_in[7];
    const float* W_hh  = (const float*)d_in[8];
    const float* b_hh  = (const float*)d_in[9];
    const float* Wproj = (const float*)d_in[10];
    const float* bproj = (const float*)d_in[11];
    float* out = (float*)d_out;

    // Workspace carve (~51 MB total), 256B aligned regions.
    char* p = (char*)d_ws;
    auto carve = [&](size_t bytes) -> char* {
        char* r = p; p += (bytes + 255) & ~(size_t)255; return r;
    };
    __bf16*   Abuf  = (__bf16*)carve((size_t)NB * 2 * HH * 2);       // [32][1024]
    __bf16*   W2    = (__bf16*)carve((size_t)HH * 2 * HH * 2);       // [512][1024]
    __bf16*   Wx    = (__bf16*)carve((size_t)HH * HH * 2);           // [512][512]
    __bf16*   Wpb   = (__bf16*)carve((size_t)VV * HH * 2);           // [32000][512]
    __bf16*   xe    = (__bf16*)carve((size_t)NB * TT * HH * 2);      // [4096][512]
    float*    xproj = (float*) carve((size_t)NB * TT * HH * 4);      // [4096][512]
    __bf16*   hs    = (__bf16*)carve((size_t)NB * TT * HH * 2);      // [4096][512]
    unsigned* bar   = (unsigned*)carve(256);

    k_init<<<dim3(64), dim3(256), 0, stream>>>(es, Wp, bp, Abuf, bar);
    k_build_w2<<<dim3(2048), dim3(256), 0, stream>>>(W_ih, W_hh, W2, Wx);
    k_cvt<<<dim3(4096), dim3(256), 0, stream>>>(Wproj, Wpb, VV * HH);
    k_gather<<<dim3(8192), dim3(256), 0, stream>>>(x, emb, xe);

    // xproj = xe @ W_ih[:, :E]^T + b_ih + b_hh   (M=4096, N=512, K=512)
    k_gemm<<<dim3(HH / 128, (NB * TT) / 64), dim3(256), 0, stream>>>(
        xe, HH, Wx, HH, b_ih, b_hh, xproj, HH, HH);

    // Persistent attention+RNN scan: 32 WGs, ~139 KB LDS each (encoder slice resident).
    size_t smem = 2048 * 4 + 256 * 4 + 128 * 4 + (size_t)HH * 2 + (size_t)T1E * EPAD * 2;
    k_scan<<<dim3(NB), dim3(256), smem, stream>>>(enc, xproj, W2, Abuf, hs, bar);

    // out = hs @ Wproj^T + bproj   (M=4096, N=32000, K=512) — HBM-bound on 524 MB write.
    k_gemm<<<dim3(VV / 128, (NB * TT) / 64), dim3(256), 0, stream>>>(
        hs, HH, Wpb, HH, bproj, nullptr, out, VV, HH);
}
